// MyInterleavedModule_38336878084188
// MI455X (gfx1250) — matrix operand order repaired and checked by reading, not compiled, not run on hardware
//
#include <hip/hip_runtime.h>

// CDNA5 / gfx1250: out = x @ (W_dense + W_sparse)^T  == single dense f16 GEMM
//   x: [8192,4096] f16, W*: [4096,4096] f16 (row-major over K), out: [8192,4096] f16
//
// - 128x128 block tile, BK=32, 8 waves (wave32), wave tile 64x32,
//   8x v_wmma_f32_16x16x32_f16 per K-step.
// - A (x) tile: Tensor Data Mover DMA global->LDS with hardware row padding
//   (pad_interval = 16 DWORDs = one 64B row, pad_amount = 4 DWORDs -> 80B
//   LDS row stride), issued by wave 0 only, fenced with s_wait_tensorcnt.
// - B (weights) tile: global->VGPR, v_pk_add_f16 fold of dense+sparse, ds_store.
// - Output: non-temporal stores (write-once 64MB; keep L2 for x/W reuse).

typedef __attribute__((ext_vector_type(16))) _Float16 v16h;
typedef __attribute__((ext_vector_type(8)))  _Float16 v8h;
typedef __attribute__((ext_vector_type(8)))  float    v8f;
typedef __attribute__((ext_vector_type(4)))  unsigned int v4u;
typedef __attribute__((ext_vector_type(8)))  int v8i;
typedef __attribute__((ext_vector_type(4)))  int v4i;

#define IN_F   4096
#define OUT_F  4096
#define NTOK   8192

#define BM 128
#define BN 128
#define BK 32
#define LDST 40                      // LDS row stride in halfs (32 + 8 pad = 80 B)

#define BUF_HALFS (BM * LDST)        // 5120 halfs = 10240 B per buffer
#define ABUF(b)   ((b) * BUF_HALFS)                 // half offsets: 0 / 5120
#define BBUF(b)   (2 * BUF_HALFS + (b) * BUF_HALFS) // half offsets: 10240 / 15360
#define LDS_BYTES (4 * BUF_HALFS * 2)               // 40960 B dynamic LDS @ offset 0

// Issue a TDM DMA of one 128x32-half x tile (row-major, row stride IN_F halfs)
// into LDS at byte offset lds_byte_off, with HW padding to an 80 B row stride.
__device__ __forceinline__ void tdm_load_x_tile(const _Float16* gsrc,
                                                unsigned int lds_byte_off)
{
    unsigned long long ga = (unsigned long long)(uintptr_t)gsrc;

    // D# group 0 (ISA 8.3): count=1 | lds_addr | global_addr[56:0] | type=2
    v4u g0;
    g0.x = 1u;
    g0.y = lds_byte_off;
    g0.z = (unsigned int)ga;
    g0.w = (unsigned int)((ga >> 32) & 0x1FFFFFFull) | (2u << 30);

    // D# group 1 (ISA 8.4)
    v8i g1;
    g1[0] = (int)((1u << 16)          // data_size = 2 bytes
                | (1u << 20)          // pad_enable
                | (3u << 22)          // pad_interval: 16 DWORDs (one 64 B tile row)
                | (3u << 25));        // pad_amount: 4 DWORDs (16 B) -> 80 B stride
    g1[1] = (int)((unsigned int)IN_F << 16);   // tensor_dim0[15:0] = 4096
    g1[2] = (int)((unsigned int)NTOK << 16);   // tensor_dim1[15:0] = 8192
    g1[3] = (int)((unsigned int)BK   << 16);   // tile_dim0 = 32 halfs
    g1[4] = BM;                                // tile_dim1 = 128, tile_dim2 = 0
    g1[5] = IN_F;                              // tensor_dim0_stride[31:0] = 4096
    g1[6] = 0;
    g1[7] = 0;

    v4i gz = (v4i){};                 // groups 2/3 unused for 2-D tensors
#if __clang_major__ >= 23
    __builtin_amdgcn_tensor_load_to_lds(g0, g1, gz, gz, (v8i){}, 0);
#else
    __builtin_amdgcn_tensor_load_to_lds(g0, g1, gz, gz, 0);
#endif
}

__global__ __launch_bounds__(256)
void fused_24_gemm_wmma_tdm(const _Float16* __restrict__ X,
                            const _Float16* __restrict__ Wd,
                            const _Float16* __restrict__ Ws,
                            _Float16* __restrict__ Out)
{
    extern __shared__ _Float16 smem[];   // dynamic LDS -> starts at offset 0

    const int tid   = threadIdx.x;
    const int lane  = tid & 31;
    const int waveM = (tid >> 5) >> 2;   // 0..1 -> 64-row slab
    const int waveN = (tid >> 5) & 3;    // 0..3 -> 32-col slab

    const int nBase = blockIdx.x * BN;
    const int mBase = blockIdx.y * BM;

    // B-tile cooperative load: 2 threads per 64 B weight row segment
    const int ldRow = tid >> 1;          // 0..127
    const int ldCol = (tid & 1) * 16;    // 0 or 16 halfs

    const _Float16* wdg   = Wd + (size_t)(nBase + ldRow) * IN_F + ldCol;
    const _Float16* wsg   = Ws + (size_t)(nBase + ldRow) * IN_F + ldCol;
    const _Float16* xtile = X  + (size_t)mBase * IN_F;   // + ks*BK per step

    // Scalarized so only wave 0 issues TDM (TDM ignores EXEC; an EXEC-masked
    // "if" would still issue from all 8 waves).
    const bool tdmWave = (__builtin_amdgcn_readfirstlane(tid >> 5) == 0);

    v8f acc[4][2];
#pragma unroll
    for (int mt = 0; mt < 4; ++mt)
#pragma unroll
        for (int nt = 0; nt < 2; ++nt)
            acc[mt][nt] = (v8f){};

    // ---- prologue: DMA A tile 0, stage B tile 0 ----
    if (tdmWave)
        tdm_load_x_tile(xtile, (unsigned int)(ABUF(0) * 2));
    {
        v8h d0 = *(const v8h*)(wdg);
        v8h d1 = *(const v8h*)(wdg + 8);
        v8h s0 = *(const v8h*)(wsg);
        v8h s1 = *(const v8h*)(wsg + 8);
        *(v8h*)&smem[BBUF(0) + ldRow * LDST + ldCol]     = d0 + s0;
        *(v8h*)&smem[BBUF(0) + ldRow * LDST + ldCol + 8] = d1 + s1;
    }
    if (tdmWave)
        __builtin_amdgcn_s_wait_tensorcnt(0);
    __syncthreads();

    const int KSTEPS = IN_F / BK;   // 128
    for (int ks = 0; ks < KSTEPS; ++ks) {
        const int buf  = ks & 1;
        const bool more = (ks + 1) < KSTEPS;

        // kick next A tile DMA into the other buffer (overlaps whole K-step)
        if (more && tdmWave)
            tdm_load_x_tile(xtile + (ks + 1) * BK,
                            (unsigned int)(ABUF(buf ^ 1) * 2));

        // issue next B tile global loads early
        v8h nd0, nd1, ns0, ns1;
        if (more) {
            const int ko = (ks + 1) * BK;
            nd0 = *(const v8h*)(wdg + ko);
            nd1 = *(const v8h*)(wdg + ko + 8);
            ns0 = *(const v8h*)(wsg + ko);
            ns1 = *(const v8h*)(wsg + ko + 8);
        }

        // ---- A fragments: lane<16 holds K{0..7,16..23}, lane>=16 K{8..15,24..31}
        const int aoff = (lane < 16) ? 0 : 8;
        v16h afrag[4];
#pragma unroll
        for (int mt = 0; mt < 4; ++mt) {
            const _Float16* p =
                &smem[ABUF(buf) + (waveM * 64 + mt * 16 + (lane & 15)) * LDST + aoff];
            v8h lo = *(const v8h*)(p);
            v8h hi = *(const v8h*)(p + 16);
            afrag[mt] = __builtin_shufflevector(lo, hi,
                0,1,2,3,4,5,6,7,8,9,10,11,12,13,14,15);
        }

        // ---- B fragments: lane n = column n; lanes 0-15 K{0..7,16..23}, 16-31 K{8..15,24..31}
        const int boff = (lane < 16) ? 0 : 8;
        v16h bfrag[2];
#pragma unroll
        for (int nt = 0; nt < 2; ++nt) {
            const _Float16* p =
                &smem[BBUF(buf) + (waveN * 32 + nt * 16 + (lane & 15)) * LDST + boff];
            v8h lo = *(const v8h*)(p);
            v8h hi = *(const v8h*)(p + 16);
            bfrag[nt] = __builtin_shufflevector(lo, hi,
                0,1,2,3,4,5,6,7,8,9,10,11,12,13,14,15);
        }

        // ---- 8x v_wmma_f32_16x16x32_f16 ----
#pragma unroll
        for (int mt = 0; mt < 4; ++mt)
#pragma unroll
            for (int nt = 0; nt < 2; ++nt)
                acc[mt][nt] = __builtin_amdgcn_wmma_f32_16x16x32_f16(
                    false, afrag[mt], false, bfrag[nt],
                    (short)0, acc[mt][nt], false, false);

        // ---- stage next B tile (fold dense+sparse) ----
        if (more) {
            const int nb = buf ^ 1;
            *(v8h*)&smem[BBUF(nb) + ldRow * LDST + ldCol]     = nd0 + ns0;
            *(v8h*)&smem[BBUF(nb) + ldRow * LDST + ldCol + 8] = nd1 + ns1;
        }

        if (tdmWave)
            __builtin_amdgcn_s_wait_tensorcnt(0);
        __syncthreads();
    }

    // ---- epilogue: lane<16 -> M=r, lane>=16 -> M=r+8; N = lane&15 ----
    const int col = nBase + waveN * 32 + (lane & 15);
#pragma unroll
    for (int mt = 0; mt < 4; ++mt) {
#pragma unroll
        for (int nt = 0; nt < 2; ++nt) {
#pragma unroll
            for (int r = 0; r < 8; ++r) {
                const int row = mBase + waveM * 64 + mt * 16 +
                                ((lane < 16) ? r : r + 8);
                __builtin_nontemporal_store(
                    (_Float16)acc[mt][nt][r],
                    &Out[(size_t)row * OUT_F + col + nt * 16]);
            }
        }
    }
}

extern "C" void kernel_launch(void* const* d_in, const int* in_sizes, int n_in,
                              void* d_out, int out_size, void* d_ws, size_t ws_size,
                              hipStream_t stream) {
    (void)in_sizes; (void)n_in; (void)out_size; (void)d_ws; (void)ws_size;
    const _Float16* x  = (const _Float16*)d_in[0];
    const _Float16* wd = (const _Float16*)d_in[1];
    const _Float16* ws = (const _Float16*)d_in[2];
    _Float16* out = (_Float16*)d_out;

    dim3 grid(OUT_F / BN, NTOK / BM);   // (32, 64)
    dim3 block(256);
    hipLaunchKernelGGL(fused_24_gemm_wmma_tdm, grid, block, LDS_BYTES, stream,
                       x, wd, ws, out);
}
